// Additive_Attention_24816321036435
// MI455X (gfx1250) — matrix operand order repaired and checked by reading, not compiled, not run on hardware
//
#include <hip/hip_runtime.h>
#include <math.h>

// Problem sizes (fixed by the reference)
#define S_LEN   1024
#define B_SZ    64
#define DDEC    1024
#define DALIGN  1024
#define DENC    1024

typedef __attribute__((ext_vector_type(16))) __bf16 v16bf;
typedef __attribute__((ext_vector_type(8)))  __bf16 v8bf;
typedef __attribute__((ext_vector_type(8)))  float  v8f;

// hardware tanh if the gfx1250 builtin is available, else libm
#if __has_builtin(__builtin_amdgcn_tanhf)
#define FAST_TANH(x) __builtin_amdgcn_tanhf(x)
#else
#define FAST_TANH(x) tanhf(x)
#endif

// fp32 -> bf16 bits, round-to-nearest-even
static __device__ __forceinline__ unsigned short bf_bits(float x) {
    unsigned u = __builtin_bit_cast(unsigned, x);
    u += 0x7FFFu + ((u >> 16) & 1u);
    return (unsigned short)(u >> 16);
}

// ---------------------------------------------------------------------------
// Kernel 0: fp32 -> bf16 pre-conversion (s_tm1: 64K elems, sa_w: 1M elems).
// float4 in, ushort4 out. Streams 4.4 MB total -- negligible vs 536 MB floor,
// and removes all conversion VALU from the WMMA inner loop.
// ---------------------------------------------------------------------------
__global__ void cvt_bf16(const float* __restrict__ in,
                         unsigned short* __restrict__ out, int n4) {
    const int i = blockIdx.x * blockDim.x + threadIdx.x;
    if (i < n4) {
        const float4 f = ((const float4*)in)[i];
        ushort4 h;
        h.x = bf_bits(f.x);
        h.y = bf_bits(f.y);
        h.z = bf_bits(f.z);
        h.w = bf_bits(f.w);
        ((ushort4*)out)[i] = h;
    }
}

// ---------------------------------------------------------------------------
// Kernel 1: sa_s[b,a] = sum_d s_tm1[b,d] * sa_w[a,d] + sa_b[a]
// 64x1024 output, K=1024, v_wmma_f32_16x16x32_bf16.
// One wave32 per 16x16 tile; 256 waves = 64 blocks x 128 threads.
// Inner loop: 4x global_load_b128 + 1 wmma per 32-K step.
// Waves 4i..4i+3 share tileN -> B fragments hit in cache.
// ---------------------------------------------------------------------------
__global__ void sa_gemm_wmma(const __bf16* __restrict__ s_bf,   // [64][1024] bf16
                             const __bf16* __restrict__ w_bf,   // [1024][1024] bf16
                             const float*  __restrict__ sa_b,
                             float* __restrict__ sa_s) {
    const int wid   = (blockIdx.x * blockDim.x + threadIdx.x) >> 5;
    const int lane  = threadIdx.x & 31;
    const int tileM = wid & 3;    // 4 tiles over B=64 rows
    const int tileN = wid >> 2;   // 64 tiles over DALIGN cols
    const int half  = lane >> 4;  // 0: lanes 0-15, 1: lanes 16-31
    const int l15   = lane & 15;

    const __bf16* arow = s_bf + (size_t)(tileM * 16 + l15) * DDEC;  // A row
    const __bf16* brow = w_bf + (size_t)(tileN * 16 + l15) * DDEC;  // B[k][n]=sa_w[n][k]

    union frag { v16bf v; struct { v8bf lo, hi; } p; };

    v8f acc = {};
    for (int k0 = 0; k0 < DDEC; k0 += 32) {
        frag a, b;
        const int ka = k0 + half * 8;    // A-frag: k = ka..ka+7, ka+16..ka+23
        const int kb = k0 + half * 8;    // B-frag: k = kb..kb+7, kb+16..kb+23
        a.p.lo = *(const v8bf*)(arow + ka);        // 16B aligned
        a.p.hi = *(const v8bf*)(arow + ka + 16);
        b.p.lo = *(const v8bf*)(brow + kb);        // 16B aligned
        b.p.hi = *(const v8bf*)(brow + kb + 16);
        acc = __builtin_amdgcn_wmma_f32_16x16x32_bf16(
            /*neg_a=*/false, a.v, /*neg_b=*/false, b.v,
            /*c_mod=*/(short)0, acc, /*reuse_a=*/false, /*reuse_b=*/false);
    }

    const int   colB  = tileN * 16 + l15;
    const float bias  = sa_b[colB];
    const int   mbase = tileM * 16 + half * 8;
#pragma unroll
    for (int r = 0; r < 8; ++r)
        sa_s[(size_t)(mbase + r) * DALIGN + colB] = acc[r] + bias;
}

// ---------------------------------------------------------------------------
// Kernel 2: e[b][s] = sum_a tanh(sa_s[b,a] + uh[s,b,a]) * a1_w[a] + a1_b
// One wave per (s,b): 65536 waves streaming 4KB of uh each via b128 loads.
// Carries the 268 MB uh read -> ~11.5 us at 23.3 TB/s; tanh is one HW trans
// op per element and co-executes with the FMAs.
// ---------------------------------------------------------------------------
__global__ void energy_kernel(const float* __restrict__ sa_s,
                              const float* __restrict__ uh,
                              const float* __restrict__ a1_w,
                              const float* __restrict__ a1_b,
                              float* __restrict__ e_raw /* [B][S] */) {
    const int wid  = blockIdx.x * (blockDim.x >> 5) + (threadIdx.x >> 5);
    const int lane = threadIdx.x & 31;
    const int b    = wid & (B_SZ - 1);
    const int s    = wid >> 6;

    const float4* urow = (const float4*)(uh + ((size_t)s * B_SZ + b) * (size_t)DALIGN);
    const float4* srow = (const float4*)(sa_s + (size_t)b * DALIGN);
    const float4* wrow = (const float4*)a1_w;

    float acc = 0.f;
#pragma unroll
    for (int i = 0; i < DALIGN / 128; ++i) {   // 8 iterations, 512B/wave each
        const int idx = i * 32 + lane;
        const float4 u4 = urow[idx];
        const float4 s4 = srow[idx];
        const float4 w4 = wrow[idx];
        acc += FAST_TANH(s4.x + u4.x) * w4.x;
        acc += FAST_TANH(s4.y + u4.y) * w4.y;
        acc += FAST_TANH(s4.z + u4.z) * w4.z;
        acc += FAST_TANH(s4.w + u4.w) * w4.w;
    }
#pragma unroll
    for (int off = 16; off; off >>= 1) acc += __shfl_xor(acc, off, 32);
    if (lane == 0) e_raw[b * S_LEN + s] = acc + a1_b[0];
}

// ---------------------------------------------------------------------------
// Kernel 3: masked softmax over S per batch column. One block per b.
// Normalizes e in place ([B][S] for coalesced downstream reads) and writes
// the [S][B] layout the output contract requires.
// ---------------------------------------------------------------------------
__global__ void softmax_kernel(const float* __restrict__ xs_mask,
                               float* __restrict__ e,     /* [B][S] in/out */
                               float* __restrict__ e_out  /* d_out, [S][B] */) {
    __shared__ float red[256];
    const int b = blockIdx.x;
    const int t = threadIdx.x;

    float v[4];
    float mx = -INFINITY;
#pragma unroll
    for (int j = 0; j < 4; ++j) {
        v[j] = e[b * S_LEN + t + j * 256];
        mx = fmaxf(mx, v[j]);
    }
    red[t] = mx;
    __syncthreads();
    for (int w = 128; w; w >>= 1) {
        if (t < w) red[t] = fmaxf(red[t], red[t + w]);
        __syncthreads();
    }
    mx = red[0];
    __syncthreads();

    float sum = 0.f;
#pragma unroll
    for (int j = 0; j < 4; ++j) {
        const int s = t + j * 256;
        v[j] = __expf(v[j] - mx) * xs_mask[s * B_SZ + b];
        sum += v[j];
    }
    red[t] = sum;
    __syncthreads();
    for (int w = 128; w; w >>= 1) {
        if (t < w) red[t] += red[t + w];
        __syncthreads();
    }
    const float inv = 1.0f / red[0];
#pragma unroll
    for (int j = 0; j < 4; ++j) {
        const int   s   = t + j * 256;
        const float val = v[j] * inv;
        e[b * S_LEN + s]    = val;   // [B][S] for attend kernel
        e_out[s * B_SZ + b] = val;   // e_ij output, [S][B]
    }
}

// ---------------------------------------------------------------------------
// Kernel 4: partial attend.  attend[b,d] = sum_s w[s]*xs_h[s,b,d], S split
// into 4 segments for occupancy (1024 blocks). Carries the 268 MB xs_h read.
// ---------------------------------------------------------------------------
__global__ void attend_partial(const float* __restrict__ e_norm, /* [B][S] */
                               const float* __restrict__ xs_h,
                               float* __restrict__ part /* [4][B][DENC] */) {
    __shared__ float wsh[256];
    const int t      = threadIdx.x;
    const int dchunk = blockIdx.x & 3;
    const int b      = (blockIdx.x >> 2) & (B_SZ - 1);
    const int sseg   = blockIdx.x >> 8;
    const int d      = dchunk * 256 + t;

    wsh[t] = e_norm[b * S_LEN + sseg * 256 + t];
    __syncthreads();

    float acc = 0.f;
    const float* base = xs_h + ((size_t)(sseg * 256) * B_SZ + b) * (size_t)DENC + d;
#pragma unroll 4
    for (int i = 0; i < 256; ++i)
        acc += wsh[i] * base[(size_t)i * (B_SZ * DENC)];

    part[((size_t)sseg * B_SZ + b) * DENC + d] = acc;
}

// Kernel 5: sum the 4 S-segment partials into the attend output region.
__global__ void attend_reduce(const float* __restrict__ part,
                              float* __restrict__ out /* d_out + S*B */) {
    const int idx = blockIdx.x * blockDim.x + threadIdx.x;  // 0..65535
    out[idx] = part[idx] + part[idx + 65536] + part[idx + 131072] + part[idx + 196608];
}

// ---------------------------------------------------------------------------
extern "C" void kernel_launch(void* const* d_in, const int* in_sizes, int n_in,
                              void* d_out, int out_size, void* d_ws, size_t ws_size,
                              hipStream_t stream) {
    const float* s_tm1   = (const float*)d_in[0];
    const float* xs_h    = (const float*)d_in[1];
    const float* uh      = (const float*)d_in[2];
    const float* xs_mask = (const float*)d_in[3];
    const float* sa_w    = (const float*)d_in[4];
    const float* sa_b    = (const float*)d_in[5];
    const float* a1_w    = (const float*)d_in[6];
    const float* a1_b    = (const float*)d_in[7];

    float* out = (float*)d_out;             // [S*B e_ij | B*DENC attend]
    float* ws  = (float*)d_ws;

    float* sa_s = ws;                       // 65536 floats
    float* e    = ws + 65536;               // 65536 floats, [B][S]
    float* part = ws + 131072;              // 262144 floats, [4][B][DENC]
    unsigned short* s_bf = (unsigned short*)(ws + 393216);   // 65536 bf16
    unsigned short* w_bf = s_bf + 65536;                     // 1048576 bf16

    // fp32 -> bf16 staging for the WMMA GEMM
    cvt_bf16<<<  64, 256, 0, stream>>>(s_tm1, s_bf, 65536 / 4);
    cvt_bf16<<<1024, 256, 0, stream>>>(sa_w,  w_bf, 1048576 / 4);

    sa_gemm_wmma  <<<  64, 128, 0, stream>>>((const __bf16*)s_bf, (const __bf16*)w_bf,
                                             sa_b, sa_s);
    energy_kernel <<<8192, 256, 0, stream>>>(sa_s, uh, a1_w, a1_b, e);
    softmax_kernel<<<  64, 256, 0, stream>>>(xs_mask, e, out);
    attend_partial<<<1024, 256, 0, stream>>>(e, xs_h, part);
    attend_reduce <<< 256, 256, 0, stream>>>(part, out + S_LEN * B_SZ);
}
